// FCGATLayer_75866302317035
// MI455X (gfx1250) — matrix operand order repaired and checked by reading, not compiled, not run on hardware
//
#include <hip/hip_runtime.h>

typedef __attribute__((ext_vector_type(16))) _Float16 v16h;
typedef __attribute__((ext_vector_type(8)))  float    v8f;
typedef int v4i __attribute__((vector_size(16)));

#define DIMS 256
#define NNODE 128
#define NEGINF (-4294967295.0f)

// ---------------------------------------------------------------------------
// WMMA fragment layouts (CDNA5 ISA 7.12.2, wave32):
//  A (16x32, MxK, f16): lane = (m&15) + 16*hi ; within the lane's 16 halfs
//     e = r*2+p, k = (r>>2)*16 + hi*8 + (r&3)*2 + p
//  B (32x16, KxN, f16): lane = (n&15) + 16*hk ; k = (r>>2)*16 + hk*8 + (r&3)*2 + p
//     (as A: lanes 0-15 hold K=0-7,16-23, lanes 16-31 hold K=8-15,24-31)
//  C/D (16x16 f32): vgpr r -> row r + 8*(lane>>4), col = lane&15
// Both A panels (LDS) and B weights (global) are stored "fragment-major":
//   flat = (tile*32 + lane)*16 + e  -> one contiguous 32B v16h load per lane.
// ---------------------------------------------------------------------------

__device__ __forceinline__ int swz_pos_a(int m, int k) {
  int kl = k & 31;
  int hi = (kl >> 3) & 1;
  int g  = kl >> 4;
  int q  = kl & 7;
  int lane = (m & 15) + (hi << 4);
  int e = (((g << 2) + (q >> 1)) << 1) + (q & 1);
  return lane * 16 + e;
}

__device__ __forceinline__ v8f wmma_f16(v16h a, v16h b, v8f c) {
  return __builtin_amdgcn_wmma_f32_16x16x32_f16(false, a, false, b, (short)0, c,
                                                false, false);
}

// ---------------------------------------------------------------------------
// Async global->LDS staging (GLOBAL_LOAD_ASYNC_TO_LDS_B128, ASYNCcnt).
// Builtin signature (from hipcc diagnostic): (int4 AS1*, int4 AS3*, Imm, Imm)
// Guarded: falls back to plain loads if the builtin is unavailable.
// ---------------------------------------------------------------------------
#if defined(__HIP_DEVICE_COMPILE__) && \
    __has_builtin(__builtin_amdgcn_global_load_async_to_lds_b128)
#define HAVE_ASYNC 1
__device__ __forceinline__ void async_copy16(const float* g, float* l) {
  __builtin_amdgcn_global_load_async_to_lds_b128(
      (__attribute__((address_space(1))) v4i*)g,
      (__attribute__((address_space(3))) v4i*)l, 0, 0);
}
__device__ __forceinline__ void async_wait0() {
#if __has_builtin(__builtin_amdgcn_s_wait_asynccnt)
  __builtin_amdgcn_s_wait_asynccnt(0);
#else
  asm volatile("s_wait_asynccnt 0" ::: "memory");
#endif
}
#else
#define HAVE_ASYNC 0
#endif

// ---------------------------------------------------------------------------
// Kernel 0: pre-swizzle weights (fp32 -> f16 B-fragment-major) and fold
// att_edge into W_e:  we_att[k][h] = sum_d W_e[k][32h+d]*att_edge[h][d]
// ---------------------------------------------------------------------------
__device__ __forceinline__ void swz_weight(const float* __restrict__ W,
                                           _Float16* __restrict__ dst, int KST,
                                           int tid, int nthr) {
  int total = KST * 16 * 512;  // KST k-steps * 16 n-tiles * 512 elems/tile
  for (int idx = tid; idx < total; idx += nthr) {
    int e    = idx & 15;
    int lane = (idx >> 4) & 31;
    int tile = idx >> 9;
    int kt = tile % KST;
    int nt = tile / KST;
    int k = kt * 32 + ((e >> 3) << 4) + ((lane >> 4) << 3) + (e & 7);  // B layout: table order, matches A
    int n = (nt << 4) + (lane & 15);
    dst[idx] = (_Float16)W[k * DIMS + n];
  }
}

__global__ __launch_bounds__(256) void prep_kernel(
    const float* __restrict__ Wv, const float* __restrict__ We,
    const float* __restrict__ W1, const float* __restrict__ W2,
    const float* __restrict__ att_edge, _Float16* __restrict__ wv_swz,
    _Float16* __restrict__ we_swz, _Float16* __restrict__ w1_swz,
    _Float16* __restrict__ w2_swz, float* __restrict__ we_att) {
  int tid = blockIdx.x * blockDim.x + threadIdx.x;
  int nthr = gridDim.x * blockDim.x;
  swz_weight(Wv, wv_swz, 8, tid, nthr);
  swz_weight(W1, w1_swz, 8, tid, nthr);
  swz_weight(W2, w2_swz, 8, tid, nthr);
  swz_weight(We, we_swz, 2, tid, nthr);
  for (int t = tid; t < 64 * 8; t += nthr) {
    int k = t >> 3, h = t & 7;
    float a = 0.f;
#pragma unroll
    for (int d = 0; d < 32; ++d)
      a += We[k * DIMS + h * 32 + d] * att_edge[h * 32 + d];
    we_att[t] = a;  // [k*8+h]
  }
}

// ---------------------------------------------------------------------------
// Kernel 1: x_proj = node_feats @ W_v   (2048x256 @ 256x256, f16 WMMA)
// ---------------------------------------------------------------------------
__global__ __launch_bounds__(256) void xproj_kernel(
    const float* __restrict__ node, const _Float16* __restrict__ wv_swz,
    float* __restrict__ x_proj) {
  __shared__ __align__(32) _Float16 a1[16 * DIMS];
  const int tid = threadIdx.x, wave = tid >> 5, lane = tid & 31;
  const int row0 = blockIdx.x * 16;
  const float* nrow = node + (size_t)row0 * DIMS;

  for (int t = tid; t < 16 * DIMS; t += 256) {
    int m = t >> 8, k = t & 255;
    a1[(k >> 5) * 512 + swz_pos_a(m, k)] = (_Float16)nrow[t];
  }
  __syncthreads();

  const int lo = lane & 15, hi = lane >> 4;
#pragma unroll
  for (int tt = 0; tt < 2; ++tt) {
    int nt = wave * 2 + tt;
    v8f acc = {};
#pragma unroll
    for (int kk = 0; kk < 8; ++kk) {
      v16h a = *(const v16h*)&a1[(kk * 32 + lane) * 16];
      v16h b = *(const v16h*)&wv_swz[((nt * 8 + kk) * 32 + lane) * 16];
      acc = wmma_f16(a, b, acc);
    }
    int col = nt * 16 + lo;
#pragma unroll
    for (int r = 0; r < 8; ++r)
      x_proj[(size_t)(row0 + r + 8 * hi) * DIMS + col] = acc[r];
  }
}

// ---------------------------------------------------------------------------
// Kernel 2: attention. One block per (b,i).
// ---------------------------------------------------------------------------
__global__ __launch_bounds__(256) void attn_kernel(
    const float* __restrict__ node, const float* __restrict__ edge,
    const unsigned char* __restrict__ mask, const float* __restrict__ x_proj,
    const _Float16* __restrict__ we_swz, const float* __restrict__ we_att,
    const float* __restrict__ att_src, const float* __restrict__ att_dst,
    const float* __restrict__ bias_node, const float* __restrict__ bias_edge,
    const float* __restrict__ ln1_g, const float* __restrict__ ln1_b,
    float* __restrict__ h1) {
  __shared__ float ef32[NNODE * 64];                  // 32 KB edge tile fp32
  __shared__ __align__(32) _Float16 ef16[NNODE * 64]; // 16 KB f16 A-fragments
  __shared__ float s_att[NNODE * 8];
  __shared__ float s_dst[8];
  __shared__ float s_part[8 * DIMS];
  __shared__ float s_red[16];

  const int tid = threadIdx.x, wave = tid >> 5, lane = tid & 31;
  const int bi = blockIdx.x;  // b*128 + i
  const int b = bi >> 7, i = bi & 127;

  const float* erow = edge + (size_t)bi * (NNODE * 64);
  const float* xb = x_proj + (size_t)b * (NNODE * DIMS);

#if HAVE_ASYNC
#pragma unroll
  for (int step = 0; step < 8; ++step) {      // 8 x (256 thr x 16B) = 32 KB
    int f4 = (step * 256 + tid) * 4;
    async_copy16(erow + f4, &ef32[f4]);
  }
#else
  for (int t = tid; t < NNODE * 64; t += 256) ef32[t] = erow[t];
#endif
  if (tid < 8) {
    float a = 0.f;
    const float* xr = xb + i * DIMS + tid * 32;
    const float* wd = att_dst + tid * 32;
#pragma unroll
    for (int d = 0; d < 32; ++d) a += xr[d] * wd[d];
    s_dst[tid] = a;
  }
#if HAVE_ASYNC
  async_wait0();
#endif
  __syncthreads();

  const unsigned char* mrow = mask + (size_t)bi * NNODE;
  for (int t = tid; t < NNODE * 8; t += 256) {
    int j = t >> 3, h = t & 7;
    float ea = 0.f;
    const float* er = &ef32[j * 64];
#pragma unroll 8
    for (int k = 0; k < 64; ++k) ea += er[k] * we_att[k * 8 + h];
    float sv = 0.f;
    const float* xr = xb + j * DIMS + h * 32;
    const float* wsv = att_src + h * 32;
#pragma unroll
    for (int d = 0; d < 32; ++d) sv += xr[d] * wsv[d];
    float raw = s_dst[h] + sv + ea;
    float lr = raw > 0.f ? raw : 0.1f * raw;  // leaky_relu slope 0.1
    s_att[j * 8 + h] = mrow[j] ? lr : NEGINF;
  }
  for (int t = tid; t < NNODE * 64; t += 256) {  // fp32 -> f16 A-swizzled
    int j = t >> 6, k = t & 63;
    int tile = (j >> 4) * 2 + (k >> 5);
    ef16[tile * 512 + swz_pos_a(j, k)] = (_Float16)ef32[t];
  }
  __syncthreads();

  {  // softmax over j; wave w handles head h = w, 4 rows per lane
    int h = wave;
    float v0 = s_att[(lane) * 8 + h];
    float v1 = s_att[(lane + 32) * 8 + h];
    float v2 = s_att[(lane + 64) * 8 + h];
    float v3 = s_att[(lane + 96) * 8 + h];
    float mx = fmaxf(fmaxf(v0, v1), fmaxf(v2, v3));
#pragma unroll
    for (int m = 16; m >= 1; m >>= 1) mx = fmaxf(mx, __shfl_xor(mx, m));
    float e0 = __expf(v0 - mx), e1 = __expf(v1 - mx);
    float e2 = __expf(v2 - mx), e3 = __expf(v3 - mx);
    float sm = e0 + e1 + e2 + e3;
#pragma unroll
    for (int m = 16; m >= 1; m >>= 1) sm += __shfl_xor(sm, m);
    float inv = 1.f / sm;
    s_att[(lane) * 8 + h] = e0 * inv;
    s_att[(lane + 32) * 8 + h] = e1 * inv;
    s_att[(lane + 64) * 8 + h] = e2 * inv;
    s_att[(lane + 96) * 8 + h] = e3 * inv;
  }
  __syncthreads();

  {  // WMMA e_value, wave w owns j-rows 16w..16w+15; contract vs att inline
    v16h A0 = *(const v16h*)&ef16[((wave * 2 + 0) * 32 + lane) * 16];
    v16h A1 = *(const v16h*)&ef16[((wave * 2 + 1) * 32 + lane) * 16];
    const int hi = lane >> 4;
#pragma unroll
    for (int t = 0; t < 16; ++t) {
      v8f acc = {};
      v16h B0 = *(const v16h*)&we_swz[((t * 2 + 0) * 32 + lane) * 16];
      acc = wmma_f16(A0, B0, acc);
      v16h B1 = *(const v16h*)&we_swz[((t * 2 + 1) * 32 + lane) * 16];
      acc = wmma_f16(A1, B1, acc);
      int h = t >> 1;
      float p = 0.f;
#pragma unroll
      for (int r = 0; r < 8; ++r) {
        int j = wave * 16 + r + 8 * hi;
        p += s_att[j * 8 + h] * acc[r];
      }
      p += __shfl_xor(p, 16);  // lanes lo / lo+16 hold the same column
      if (lane < 16) s_part[wave * DIMS + t * 16 + lane] = p;
    }
  }

  // att @ x_proj term: column-per-thread, coalesced row sweeps (L2 resident)
  float ov = 0.f;
  {
    int h = tid >> 5;
    const float* xc = xb + tid;
#pragma unroll 4
    for (int j = 0; j < NNODE; ++j) ov += s_att[j * 8 + h] * xc[j * DIMS];
  }
  __syncthreads();

  float val = ov;
#pragma unroll
  for (int w = 0; w < 8; ++w) val += s_part[w * DIMS + tid];
  // sum_j att == 1 -> bias_edge enters with coefficient 1
  val += bias_edge[tid] + bias_node[tid] + node[(size_t)bi * DIMS + tid];

  // LN1 over the 256-vector
  float sx = val, sx2 = val * val;
#pragma unroll
  for (int m = 16; m >= 1; m >>= 1) {
    sx += __shfl_xor(sx, m);
    sx2 += __shfl_xor(sx2, m);
  }
  if (lane == 0) { s_red[wave] = sx; s_red[8 + wave] = sx2; }
  __syncthreads();
  float tx = 0.f, tx2 = 0.f;
#pragma unroll
  for (int w = 0; w < 8; ++w) { tx += s_red[w]; tx2 += s_red[8 + w]; }
  float mu = tx * (1.f / 256.f);
  float var = tx2 * (1.f / 256.f) - mu * mu;
  float rs = rsqrtf(var + 1e-5f);
  h1[(size_t)bi * DIMS + tid] = (val - mu) * rs * ln1_g[tid] + ln1_b[tid];
}

// ---------------------------------------------------------------------------
// Kernel 3: FFN: out = LN2(h1 + relu(h1@W1+b1)@W2+b2); one 16-row tile/block
// ---------------------------------------------------------------------------
__global__ __launch_bounds__(256) void ffn_kernel(
    const float* __restrict__ h1, const _Float16* __restrict__ w1_swz,
    const _Float16* __restrict__ w2_swz, const float* __restrict__ fb1,
    const float* __restrict__ fb2, const float* __restrict__ ln2_g,
    const float* __restrict__ ln2_b, float* __restrict__ out) {
  __shared__ __align__(32) _Float16 a1[16 * DIMS];
  __shared__ __align__(32) _Float16 a2[16 * DIMS];
  __shared__ float s_o[16 * DIMS];  // doubles as fp32 h1 staging buffer
  const int tid = threadIdx.x, wave = tid >> 5, lane = tid & 31;
  const int row0 = blockIdx.x * 16;
  const float* hrow = h1 + (size_t)row0 * DIMS;

#if HAVE_ASYNC
#pragma unroll
  for (int step = 0; step < 4; ++step) {  // 16 KB h1 tile -> LDS via async DMA
    int f4 = (step * 256 + tid) * 4;
    async_copy16(hrow + f4, &s_o[f4]);
  }
  async_wait0();
  __syncthreads();
  for (int t = tid; t < 16 * DIMS; t += 256) {
    int m = t >> 8, k = t & 255;
    a1[(k >> 5) * 512 + swz_pos_a(m, k)] = (_Float16)s_o[t];
  }
#else
  for (int t = tid; t < 16 * DIMS; t += 256) {
    int m = t >> 8, k = t & 255;
    a1[(k >> 5) * 512 + swz_pos_a(m, k)] = (_Float16)hrow[t];
    s_o[t] = hrow[t];  // keep residual copy in LDS
  }
#endif
  __syncthreads();

  const int lo = lane & 15, hi = lane >> 4;
#pragma unroll
  for (int tt = 0; tt < 2; ++tt) {  // matmul1 + bias + relu -> a2 (A-swizzled)
    int nt = wave * 2 + tt;
    v8f acc = {};
#pragma unroll
    for (int kk = 0; kk < 8; ++kk) {
      v16h a = *(const v16h*)&a1[(kk * 32 + lane) * 16];
      v16h b = *(const v16h*)&w1_swz[((nt * 8 + kk) * 32 + lane) * 16];
      acc = wmma_f16(a, b, acc);
    }
    int col = nt * 16 + lo;
    float bb = fb1[col];
#pragma unroll
    for (int r = 0; r < 8; ++r) {
      int m = r + 8 * hi;
      float v = acc[r] + bb;
      v = v > 0.f ? v : 0.f;
      a2[(col >> 5) * 512 + swz_pos_a(m, col)] = (_Float16)v;
    }
  }
  __syncthreads();

#pragma unroll
  for (int tt = 0; tt < 2; ++tt) {  // matmul2 + bias + residual (from LDS)
    int nt = wave * 2 + tt;
    v8f acc = {};
#pragma unroll
    for (int kk = 0; kk < 8; ++kk) {
      v16h a = *(const v16h*)&a2[(kk * 32 + lane) * 16];
      v16h b = *(const v16h*)&w2_swz[((nt * 8 + kk) * 32 + lane) * 16];
      acc = wmma_f16(a, b, acc);
    }
    int col = nt * 16 + lo;
    float bb = fb2[col];
#pragma unroll
    for (int r = 0; r < 8; ++r) {
      int m = r + 8 * hi;
      s_o[m * DIMS + col] = acc[r] + bb + s_o[m * DIMS + col];
    }
  }
  __syncthreads();

  // LN2: 16 threads per row (aligned inside a wave -> width-16 shuffles)
  int rr = tid >> 4, c0 = tid & 15;
  float sx = 0.f, sx2 = 0.f;
  for (int c = c0; c < DIMS; c += 16) {
    float v = s_o[rr * DIMS + c];
    sx += v; sx2 += v * v;
  }
#pragma unroll
  for (int m = 8; m >= 1; m >>= 1) {
    sx += __shfl_xor(sx, m, 16);
    sx2 += __shfl_xor(sx2, m, 16);
  }
  float mu = sx * (1.f / 256.f);
  float var = sx2 * (1.f / 256.f) - mu * mu;
  float rs = rsqrtf(var + 1e-5f);
  for (int c = c0; c < DIMS; c += 16)
    out[(size_t)(row0 + rr) * DIMS + c] =
        (s_o[rr * DIMS + c] - mu) * rs * ln2_g[c] + ln2_b[c];
}

// ---------------------------------------------------------------------------
extern "C" void kernel_launch(void* const* d_in, const int* in_sizes, int n_in,
                              void* d_out, int out_size, void* d_ws,
                              size_t ws_size, hipStream_t stream) {
  (void)in_sizes; (void)n_in; (void)out_size; (void)ws_size;
  const float* node      = (const float*)d_in[0];
  const float* edge      = (const float*)d_in[1];
  const unsigned char* mask = (const unsigned char*)d_in[2];
  const float* Wv        = (const float*)d_in[3];
  const float* We        = (const float*)d_in[4];
  const float* att_src   = (const float*)d_in[5];
  const float* att_dst   = (const float*)d_in[6];
  const float* att_edge  = (const float*)d_in[7];
  const float* bias_node = (const float*)d_in[8];
  const float* bias_edge = (const float*)d_in[9];
  const float* ff_w1     = (const float*)d_in[10];
  const float* ff_b1     = (const float*)d_in[11];
  const float* ff_w2     = (const float*)d_in[12];
  const float* ff_b2     = (const float*)d_in[13];
  const float* ln1_g     = (const float*)d_in[14];
  const float* ln1_b     = (const float*)d_in[15];
  const float* ln2_g     = (const float*)d_in[16];
  const float* ln2_b     = (const float*)d_in[17];

  char* ws = (char*)d_ws;
  float* x_proj = (float*)ws;     ws += (size_t)2048 * 256 * sizeof(float);
  float* h1     = (float*)ws;     ws += (size_t)2048 * 256 * sizeof(float);
  _Float16* wv_swz = (_Float16*)ws; ws += (size_t)65536 * sizeof(_Float16);
  _Float16* w1_swz = (_Float16*)ws; ws += (size_t)65536 * sizeof(_Float16);
  _Float16* w2_swz = (_Float16*)ws; ws += (size_t)65536 * sizeof(_Float16);
  _Float16* we_swz = (_Float16*)ws; ws += (size_t)16384 * sizeof(_Float16);
  float* we_att = (float*)ws;     ws += (size_t)512 * sizeof(float);

  hipLaunchKernelGGL(prep_kernel, dim3(64), dim3(256), 0, stream,
                     Wv, We, ff_w1, ff_w2, att_edge,
                     wv_swz, we_swz, w1_swz, w2_swz, we_att);
  hipLaunchKernelGGL(xproj_kernel, dim3(128), dim3(256), 0, stream,
                     node, wv_swz, x_proj);
  hipLaunchKernelGGL(attn_kernel, dim3(2048), dim3(256), 0, stream,
                     node, edge, mask, x_proj, we_swz, we_att,
                     att_src, att_dst, bias_node, bias_edge,
                     ln1_g, ln1_b, h1);
  hipLaunchKernelGGL(ffn_kernel, dim3(128), dim3(256), 0, stream,
                     h1, w1_swz, w2_swz, ff_b1, ff_b2, ln2_g, ln2_b,
                     (float*)d_out);
}
